// MultiHeadGraphAttention_14989435863222
// MI455X (gfx1250) — matrix operand order repaired and checked by reading, not compiled, not run on hardware
//
#include <hip/hip_runtime.h>
#include <hip/hip_bf16.h>

// GAT forward for MI455X (gfx1250, wave32, WMMA).
// N=4096, F_IN=256, H=8, F_OUT=64, leaky slope 0.2.

typedef __attribute__((ext_vector_type(16))) _Float16 v16h;
typedef __attribute__((ext_vector_type(8)))  _Float16 v8h;
typedef __attribute__((ext_vector_type(4)))  _Float16 v4h;
typedef __attribute__((ext_vector_type(8)))  float    v8f;

#define GAT_N     4096
#define GAT_FIN   256
#define GAT_H     8
#define GAT_FOUT  64
#define NEG_SLOPE 0.2f
#define PSTRIDE   (GAT_N + 8)   // halves per LDS row (pad to avoid bank conflicts)

// ---------------------------------------------------------------------------
// Kernel 0: w[h][k][o] (f32) -> wT[h][o][k] (f16), so WMMA B-lanes read
// contiguous K.
// ---------------------------------------------------------------------------
__global__ void gat_wtrans(const float* __restrict__ w, _Float16* __restrict__ wT) {
    int idx = blockIdx.x * blockDim.x + threadIdx.x;      // < 8*64*256
    if (idx >= GAT_H * GAT_FOUT * GAT_FIN) return;
    int k  = idx & (GAT_FIN - 1);
    int o  = (idx >> 8) & (GAT_FOUT - 1);
    int hh = idx >> 14;
    wT[idx] = (_Float16)w[(hh * GAT_FIN + k) * GAT_FOUT + o];
}

// ---------------------------------------------------------------------------
// Kernel 1: h_prime = h @ w per head, via v_wmma_f32_16x16x32_f16.
// Grid (256 row-tiles, 8 heads), 128 threads = 4 waves; wave -> 16-wide o-tile.
// Writes h_prime f32 [H][N][FOUT] and h_primeT f16 [H][FOUT][N].
// ---------------------------------------------------------------------------
__global__ __launch_bounds__(128) void gat_proj(
        const float*    __restrict__ h,
        const _Float16* __restrict__ wT,
        float*          __restrict__ hp,
        _Float16*       __restrict__ hpT) {
    const int head = blockIdx.y;
    const int r0   = blockIdx.x * 16;
    const int lane = threadIdx.x & 31;
    const int wave = threadIdx.x >> 5;
    const int lm   = lane & 15;
    const int hs   = lane >> 4;          // half-wave select
    const int o0   = wave * 16;

    const float*    hrow = h  + (size_t)(r0 + lm) * GAT_FIN;
    const _Float16* brow = wT + (size_t)(head * GAT_FOUT + o0 + lm) * GAT_FIN;

    v8f acc = {};
    #pragma unroll
    for (int k = 0; k < GAT_FIN / 32; ++k) {
        // A fragment: 16-bit A 16x32 layout; lane holds row r0+lm,
        // K = base+{0..7} in elems 0..7, base+{16..23} in elems 8..15,
        // base = 8 for lanes 16..31.
        union { v16h v; _Float16 e[16]; } A;
        const int ka = k * 32 + hs * 8;
        #pragma unroll
        for (int e = 0; e < 8; ++e) A.e[e]     = (_Float16)hrow[ka + e];
        #pragma unroll
        for (int e = 0; e < 8; ++e) A.e[8 + e] = (_Float16)hrow[ka + 16 + e];
        // B fragment: lane holds column o0+lm, same K order as A
        // (base+{0..7}, base+{16..23}, base = 8 for lanes 16..31).
        union { v16h v; v8h h[2]; } B;
        B.h[0] = *(const v8h*)(brow + k * 32 + hs * 8);
        B.h[1] = *(const v8h*)(brow + k * 32 + hs * 8 + 16);
        acc = __builtin_amdgcn_wmma_f32_16x16x32_f16(
                  false, A.v, false, B.v, (short)0, acc, false, false);
    }
    // C/D layout: VGPR v holds row M = v + 8*hs, column = lane%16.
    #pragma unroll
    for (int v = 0; v < 8; ++v) {
        const int m = v + hs * 8;
        const float val = acc[v];
        hp [((size_t)head * GAT_N + r0 + m) * GAT_FOUT + o0 + lm] = val;
        hpT[((size_t)head * GAT_FOUT + o0 + lm) * GAT_N + r0 + m] = (_Float16)val;
    }
}

// ---------------------------------------------------------------------------
// Kernel 2: attn_src[h][n] = h_prime[h][n][:] . a_src[h][:], same for dst.
// One wave per (h, n) row.
// ---------------------------------------------------------------------------
__global__ __launch_bounds__(128) void gat_srcdst(
        const float* __restrict__ hp,
        const float* __restrict__ a_src,
        const float* __restrict__ a_dst,
        float* __restrict__ srcv,
        float* __restrict__ dstv) {
    const int lane = threadIdx.x & 31;
    const int row  = blockIdx.x * 4 + (threadIdx.x >> 5);   // < H*N
    const int head = row >> 12;
    const float x0 = hp[(size_t)row * GAT_FOUT + lane];
    const float x1 = hp[(size_t)row * GAT_FOUT + 32 + lane];
    float s = x0 * a_src[head * GAT_FOUT + lane] + x1 * a_src[head * GAT_FOUT + 32 + lane];
    float d = x0 * a_dst[head * GAT_FOUT + lane] + x1 * a_dst[head * GAT_FOUT + 32 + lane];
    #pragma unroll
    for (int off = 16; off >= 1; off >>= 1) {
        s += __shfl_xor(s, off);
        d += __shfl_xor(d, off);
    }
    if (lane == 0) {
        srcv[row] = s;
        dstv[row] = d;
    }
}

// ---------------------------------------------------------------------------
// Kernel 3: fused masked-softmax attention + P @ h_prime, flash-style.
// One workgroup per (head, 16-row tile); 4 waves.
//   Phase 0: async-copy this head's dst[] segment into LDS
//            (global_load_async_to_lds_b128 + s_wait_asynccnt) — it is
//            re-read 16x in phase 1, so stage it once.
//   Phase 1: masked leaky logits -> LDS (f16), per-wave row maxima.
//   Phase 2: exp(l - max) in place, row sums.
//   Phase 3: each wave owns a 16-wide o-tile; 128 WMMAs over K=4096 with
//            A=P (from LDS, 16-bit A layout) and B=h_primeT (contiguous).
// ---------------------------------------------------------------------------
__global__ __launch_bounds__(128) void gat_attn(
        const unsigned char* __restrict__ adj,
        const _Float16*      __restrict__ hpT,
        const float*         __restrict__ srcv,
        const float*         __restrict__ dstv,
        const float*         __restrict__ bias,
        float*               __restrict__ out) {
    extern __shared__ _Float16 P[];            // 16*PSTRIDE halves + 4096 f32
    float* dstc = (float*)(P + 16 * PSTRIDE);  // LDS cache of dst[head][:]
    __shared__ float wmax[4][16];
    __shared__ float wsum[4][16];
    __shared__ float gmax[16];
    __shared__ float gsum[16];

    const int head = blockIdx.y;
    const int r0   = blockIdx.x * 16;
    const int lane = threadIdx.x & 31;
    const int wave = threadIdx.x >> 5;
    const int seg  = wave * (GAT_N / 4);       // 1024-wide K segment per wave

    const float* dsth = dstv + head * GAT_N;
    const float* srch = srcv + head * GAT_N;

    // -------- Phase 0: async-stage dst segment into LDS ---------------------
    // Each wave copies its own 1024-float segment; each lane moves 8 x 16B.
    #pragma unroll
    for (int t = 0; t < 8; ++t) {
        const int eo = seg + t * 128 + lane * 4;
        const unsigned lds_off =
            (unsigned)(unsigned long long)(const void*)(dstc + eo);
        const unsigned long long gaddr =
            (unsigned long long)(const void*)(dsth + eo);
        asm volatile("global_load_async_to_lds_b128 %0, %1, off"
                     :: "v"(lds_off), "v"(gaddr) : "memory");
    }
    asm volatile("s_wait_asynccnt 0x0" ::: "memory");
    // Segments are produced and consumed by the same wave: no barrier needed.

    // -------- Phase 1: masked leaky-relu logits, row max --------------------
    for (int i = 0; i < 16; ++i) {
        const unsigned char* arow = adj + (size_t)(r0 + i) * GAT_N + seg;
        const float si = srch[r0 + i];
        _Float16* prow = P + i * PSTRIDE + seg;
        float lmax = -3.0e38f;
        #pragma unroll 2
        for (int it = 0; it < 8; ++it) {
            const int m = it * 128 + lane * 4;
            __builtin_prefetch(arow + m + GAT_N, 0, 1);   // global_prefetch_b8
            const unsigned int ab = *(const unsigned int*)(arow + m);
            const float4 d4 = *(const float4*)(dstc + seg + m);   // LDS hit
            float l[4];
            l[0] = si + d4.x; l[1] = si + d4.y; l[2] = si + d4.z; l[3] = si + d4.w;
            #pragma unroll
            for (int j = 0; j < 4; ++j) {
                float a = l[j];
                a = (a >= 0.0f) ? a : NEG_SLOPE * a;              // leaky relu
                a = ((ab >> (8 * j)) & 0xFF) ? a : -1.0e30f;      // adjacency mask
                l[j] = a;
                lmax = fmaxf(lmax, a);
            }
            v4h st = { (_Float16)l[0], (_Float16)l[1], (_Float16)l[2], (_Float16)l[3] };
            *(v4h*)(prow + m) = st;
        }
        #pragma unroll
        for (int off = 16; off >= 1; off >>= 1) lmax = fmaxf(lmax, __shfl_xor(lmax, off));
        if (lane == 0) wmax[wave][i] = lmax;
    }
    __syncthreads();
    if (threadIdx.x < 16) {
        const int i = threadIdx.x;
        gmax[i] = fmaxf(fmaxf(wmax[0][i], wmax[1][i]), fmaxf(wmax[2][i], wmax[3][i]));
    }
    __syncthreads();

    // -------- Phase 2: exponentiate in place, row sums ----------------------
    for (int i = 0; i < 16; ++i) {
        const float gm = gmax[i];
        _Float16* prow = P + i * PSTRIDE + seg;
        float lsum = 0.0f;
        #pragma unroll 2
        for (int it = 0; it < 8; ++it) {
            const int m = it * 128 + lane * 4;
            v4h v = *(v4h*)(prow + m);
            float e0 = __expf((float)v[0] - gm);
            float e1 = __expf((float)v[1] - gm);
            float e2 = __expf((float)v[2] - gm);
            float e3 = __expf((float)v[3] - gm);
            lsum += (e0 + e1) + (e2 + e3);
            v4h st = { (_Float16)e0, (_Float16)e1, (_Float16)e2, (_Float16)e3 };
            *(v4h*)(prow + m) = st;
        }
        #pragma unroll
        for (int off = 16; off >= 1; off >>= 1) lsum += __shfl_xor(lsum, off);
        if (lane == 0) wsum[wave][i] = lsum;
    }
    __syncthreads();
    if (threadIdx.x < 16) {
        const int i = threadIdx.x;
        gsum[i] = (wsum[0][i] + wsum[1][i]) + (wsum[2][i] + wsum[3][i]);
    }
    __syncthreads();

    // -------- Phase 3: out-tile = P @ h_prime via WMMA ----------------------
    const int lm = lane & 15;
    const int hs = lane >> 4;
    const int o0 = wave * 16;
    const _Float16* pa   = P + lm * PSTRIDE;                              // A row
    const _Float16* brow = hpT + (size_t)(head * GAT_FOUT + o0 + lm) * GAT_N; // B column

    v8f acc = {};
    for (int k = 0; k < GAT_N / 32; ++k) {
        union { v16h v; v8h h[2]; } A;
        const int ba = k * 32 + hs * 8;
        A.h[0] = *(const v8h*)(pa + ba);          // K = base .. base+7
        A.h[1] = *(const v8h*)(pa + ba + 16);     // K = base+16 .. base+23
        union { v16h v; v8h h[2]; } B;
        B.h[0] = *(const v8h*)(brow + ba);
        B.h[1] = *(const v8h*)(brow + ba + 16);
        acc = __builtin_amdgcn_wmma_f32_16x16x32_f16(
                  false, A.v, false, B.v, (short)0, acc, false, false);
    }

    const float bo = bias[o0 + lm];
    #pragma unroll
    for (int v = 0; v < 8; ++v) {
        const int m = v + hs * 8;
        const float o = acc[v] / gsum[m] + bo;
        out[((size_t)head * GAT_N + r0 + m) * GAT_FOUT + o0 + lm] = o;
    }
}

// ---------------------------------------------------------------------------
// Launch
// ---------------------------------------------------------------------------
extern "C" void kernel_launch(void* const* d_in, const int* in_sizes, int n_in,
                              void* d_out, int out_size, void* d_ws, size_t ws_size,
                              hipStream_t stream) {
    const float*         h     = (const float*)d_in[0];
    const unsigned char* adj   = (const unsigned char*)d_in[1];   // jax bool -> 1 byte
    const float*         w     = (const float*)d_in[2];
    const float*         a_src = (const float*)d_in[3];
    const float*         a_dst = (const float*)d_in[4];
    const float*         b     = (const float*)d_in[5];
    float*               out   = (float*)d_out;

    // Workspace layout (~12.5 MB total)
    char* ws = (char*)d_ws;
    _Float16* wT  = (_Float16*)ws;                        // 8*64*256 f16  = 256 KB
    float*    hp  = (float*)(ws + 262144);                // 8*4096*64 f32 = 8 MB
    _Float16* hpT = (_Float16*)(ws + 262144 + 8388608);   // 8*64*4096 f16 = 4 MB
    float*    sv  = (float*)(ws + 262144 + 8388608 + 4194304);            // 128 KB
    float*    dv  = (float*)(ws + 262144 + 8388608 + 4194304 + 131072);   // 128 KB

    // 0) transpose+convert weights
    gat_wtrans<<<dim3((GAT_H * GAT_FOUT * GAT_FIN + 255) / 256), dim3(256), 0, stream>>>(w, wT);

    // 1) projection GEMM (WMMA)
    gat_proj<<<dim3(GAT_N / 16, GAT_H), dim3(128), 0, stream>>>(h, wT, hp, hpT);

    // 2) attention scalars
    gat_srcdst<<<dim3((GAT_H * GAT_N) / 4), dim3(128), 0, stream>>>(hp, a_src, a_dst, sv, dv);

    // 3) fused masked softmax + P@V (WMMA); dynamic LDS = P (128 KB) + dst cache (16 KB)
    const size_t smem = (size_t)16 * PSTRIDE * sizeof(_Float16) + GAT_N * sizeof(float);
    gat_attn<<<dim3(GAT_N / 16, GAT_H), dim3(128), smem, stream>>>(adj, hpT, sv, dv, b, out);
}
